// UnetEncoder3D_128_3109556323116
// MI455X (gfx1250) — matrix operand order repaired and checked by reading, not compiled, not run on hardware
//
#include <hip/hip_runtime.h>

typedef __attribute__((ext_vector_type(16))) _Float16 v16h;
typedef __attribute__((ext_vector_type(8)))  float    v8f;

// ---------------------------------------------------------------------------
// Direct-to-register WMMA fragments (wave32; A and B use contiguous k order, k = 16*(lane/16) + s):
//  A (16x32 f16, MxK): lanes 0-15 -> M=lane,    K 0..15
//                      lanes16-31 -> M=lane-16, K 16..31
//    -> per lane: two contiguous 8-element runs of w[o][kc+ko .. ] .
//  B (32x16 f16, KxN): lanes 0-15 -> N=lane,    K 0..15
//                      lanes16-31 -> N=lane-16, K 16..31
//    -> per lane: 16 consecutive K of one im2col column (gathered).
//  D (16x16 f32):      VGPR r -> M = r + 8*(lane>=16), N = lane&15.
// No LDS staging: every element is consumed exactly once by the loading wave.
// ---------------------------------------------------------------------------

struct ConvP {
  const float* x;    // NCDHW activations (implicit im2col mode)
  const float* w;    // weights, row-major [O][K]
  const float* col;  // pre-materialized im2col, row-major [Nsp][K]
  float* y;          // output NCDHW
  int B, C, D, H, W;
  int O, kd, kh, kw;
  int sd, sh, sw, pd, ph, pw;
  int Do, Ho, Wo;
  int K, Nsp, tilesO, tilesN;   // tilesO = O/32, tilesN = Nsp/64 (exact)
  int fuseLrelu, precol;
};

__device__ inline v16h load_a_frag_g(const float* __restrict__ wrow, int ko) {
  // wrow points at w[o][kc]; fragment = {K ko..ko+15}
  v16h a;
  const float* s0 = wrow + ko;
  const float* s1 = wrow + ko + 8;
#pragma unroll
  for (int j = 0; j < 8; ++j) a[j] = (_Float16)s0[j];
#pragma unroll
  for (int j = 0; j < 8; ++j) a[8 + j] = (_Float16)s1[j];
  return a;
}

// One wave computes a 32(M) x 64(N) output macro-tile: 2 A-frags x 4 B-frags,
// 8 WMMAs per 32-wide K chunk.
__global__ __launch_bounds__(128) void conv_wmma_kernel(ConvP p) {
  const int lane = threadIdx.x & 31;
  const int wv   = threadIdx.x >> 5;
  const int tile = blockIdx.x * 4 + wv;
  const bool act = tile < p.tilesO * p.tilesN;  // uniform per wave
  if (!act) return;
  const int tO = tile % p.tilesO;
  const int tN = tile / p.tilesO;
  const int oBase = tO << 5;   // 32 output channels per wave
  const int nBase = tN << 6;   // 64 spatial positions per wave

  const int khw   = p.kh * p.kw;
  const int kvol  = p.kd * khw;
  const int HW    = p.H * p.W;
  const int HoWo  = p.Ho * p.Wo;
  const int DHoWo = p.Do * HoWo;

  const int lm  = lane & 15;        // M/N sub-index of this lane
  const int koA = (lane >> 4) * 16; // A fragment K sub-offset
  const int koB = (lane >> 4) * 16; // B fragment K sub-offset

  // Decode the 4 output positions this lane's B columns correspond to.
  int  z0a[4], y0a[4], x0a[4];
  long cba[4];   // (long)b * C  (channel base)
  long oba[4];   // output flat base for the epilogue
#pragma unroll
  for (int nt = 0; nt < 4; ++nt) {
    const int n = nBase + nt * 16 + lm;
    const int b = n / DHoWo;  int r = n - b * DHoWo;
    const int zo = r / HoWo;  r -= zo * HoWo;
    const int yo = r / p.Wo;
    const int xo = r - yo * p.Wo;
    z0a[nt] = zo * p.sd - p.pd;
    y0a[nt] = yo * p.sh - p.ph;
    x0a[nt] = xo * p.sw - p.pw;
    cba[nt] = (long)b * p.C;
    oba[nt] = (long)b * p.O * DHoWo + (long)zo * HoWo + yo * p.Wo + xo;
  }

  const float* wrow0 = p.w + (long)(oBase + lm) * p.K;        // A frag row, mt=0
  const float* wrow1 = p.w + (long)(oBase + 16 + lm) * p.K;   // A frag row, mt=1

  v8f acc0[4] = {{}, {}, {}, {}};
  v8f acc1[4] = {{}, {}, {}, {}};

  for (int kc = 0; kc < p.K; kc += 32) {
    // ---- A fragments: contiguous loads from [O][K] weights ----
    v16h a0 = load_a_frag_g(wrow0 + kc, koA);
    v16h a1 = load_a_frag_g(wrow1 + kc, koA);
    if (kc + 32 < p.K) {   // prefetch next weight chunk (global_prefetch_b8)
      __builtin_prefetch(wrow0 + kc + 32, 0, 1);
      __builtin_prefetch(wrow1 + kc + 32, 0, 1);
    }

    // ---- B fragments ----
    v16h bf[4];
    if (p.precol) {
#pragma unroll
      for (int nt = 0; nt < 4; ++nt) {
        const int n = nBase + nt * 16 + lm;
        const float* src = p.col + (long)n * p.K + kc + koB;
#pragma unroll
        for (int j = 0; j < 16; ++j) bf[nt][j] = (_Float16)src[j];
      }
    } else {
      // incremental k -> (c,kz,ky,kx) decode: one div pair per chunk
      int k  = kc + koB;
      int c  = k / kvol;
      int r  = k - c * kvol;
      int kz = r / khw;  r -= kz * khw;
      int ky = r / p.kw;
      int kx = r - ky * p.kw;
#pragma unroll
      for (int j = 0; j < 16; ++j) {
#pragma unroll
        for (int nt = 0; nt < 4; ++nt) {
          const int z  = z0a[nt] + kz;
          const int yy = y0a[nt] + ky;
          const int xx = x0a[nt] + kx;
          float v = 0.f;
          if (z >= 0 && z < p.D && yy >= 0 && yy < p.H && xx >= 0 && xx < p.W)
            v = p.x[((cba[nt] + c) * p.D + z) * HW + yy * p.W + xx];
          bf[nt][j] = (_Float16)v;
        }
        if (++kx == p.kw) { kx = 0; if (++ky == p.kh) { ky = 0; if (++kz == p.kd) { kz = 0; ++c; } } }
      }
    }

    // ---- 8 WMMAs: 2 M-frags x 4 N-frags ----
#pragma unroll
    for (int nt = 0; nt < 4; ++nt) {
      acc0[nt] = __builtin_amdgcn_wmma_f32_16x16x32_f16(
          false, a0, false, bf[nt], (short)0, acc0[nt], false, false);
      acc1[nt] = __builtin_amdgcn_wmma_f32_16x16x32_f16(
          false, a1, false, bf[nt], (short)0, acc1[nt], false, false);
    }
  }

  // ---- Epilogue: D layout -> NCDHW ----
  const int mSub = (lane >> 4) * 8;
#pragma unroll
  for (int nt = 0; nt < 4; ++nt) {
#pragma unroll
    for (int rI = 0; rI < 8; ++rI) {
      {
        const int o = oBase + mSub + rI;
        float v = acc0[nt][rI];
        if (p.fuseLrelu) v = (v >= 0.f) ? v : 0.2f * v;
        p.y[oba[nt] + (long)o * DHoWo] = v;
      }
      {
        const int o = oBase + 16 + mSub + rI;
        float v = acc1[nt][rI];
        if (p.fuseLrelu) v = (v >= 0.f) ? v : 0.2f * v;
        p.y[oba[nt] + (long)o * DHoWo] = v;
      }
    }
  }
}

// ---------------------------------------------------------------------------
// BatchNorm (training-mode batch stats) — per-channel reduce, then apply+LReLU
// ---------------------------------------------------------------------------
__global__ __launch_bounds__(256) void bn_reduce_kernel(
    const float* __restrict__ x, float* __restrict__ mean, float* __restrict__ var,
    int C, int B, int S) {
  const int c = blockIdx.x;
  const int tid = threadIdx.x;
  const int n = B * S;
  float s = 0.f, s2 = 0.f;
  for (int i = tid; i < n; i += 256) {
    const int b = i / S;
    const int sp = i - b * S;
    const float v = x[((long)b * C + c) * S + sp];
    s += v; s2 += v * v;
  }
  __shared__ float rs[256], rq[256];
  rs[tid] = s; rq[tid] = s2;
  __syncthreads();
  for (int o = 128; o > 0; o >>= 1) {
    if (tid < o) { rs[tid] += rs[tid + o]; rq[tid] += rq[tid + o]; }
    __syncthreads();
  }
  if (tid == 0) {
    const float inv = 1.f / (float)n;
    const float m = rs[0] * inv;
    mean[c] = m;
    var[c]  = rq[0] * inv - m * m;
  }
}

__global__ void bn_apply_lrelu_kernel(
    const float* __restrict__ x, float* __restrict__ y,
    const float* __restrict__ mean, const float* __restrict__ var,
    const float* __restrict__ g, const float* __restrict__ bta,
    int C, int S, long total) {
  const long i = (long)blockIdx.x * blockDim.x + threadIdx.x;
  if (i >= total) return;
  const int c = (int)((i / S) % C);
  float v = (x[i] - mean[c]) * rsqrtf(var[c] + 1e-5f) * g[c] + bta[c];
  y[i] = (v >= 0.f) ? v : 0.2f * v;
}

// ---------------------------------------------------------------------------
// Deformable conv3d sampling: trilinear gather -> im2col buffer [Nsp][K]
// x: enc4 [64,256,2,8,8]; off: [64,192,2,4,4] (G=4,Kk=16,comp=3 packed in ch)
// col layout: col[(b*32+pos)*4096 + c*16 + k]  (K = C*Kk = 4096)
// ---------------------------------------------------------------------------
__global__ void deform_sample_kernel(const float* __restrict__ x,
                                     const float* __restrict__ off,
                                     float* __restrict__ col) {
  const int t = blockIdx.x * blockDim.x + threadIdx.x;
  if (t >= 64 * 4 * 16 * 32) return;
  const int pos = t & 31;
  int tmp = t >> 5;
  const int k = tmp & 15; tmp >>= 4;
  const int g = tmp & 3;
  const int b = tmp >> 2;
  const int zo = pos >> 4, rem = pos & 15, yo = rem >> 2, xo = rem & 3;

  const float* ob = off + ((long)b * 192 + (g * 16 + k) * 3) * 32 + pos;
  const float pz = ob[0]  + (float)zo;                          // sd=1, pd=0, kz=0
  const float py = ob[32] + (float)(yo * 2 - 1 + (k >> 2));     // sh=2, ph=1
  const float px = ob[64] + (float)(xo * 2 - 1 + (k & 3));      // sw=2, pw=1

  const float z0f = floorf(pz), y0f = floorf(py), x0f = floorf(px);
  const float fz = pz - z0f, fy = py - y0f, fx = px - x0f;
  const float wz[2] = {1.f - fz, fz}, wy[2] = {1.f - fy, fy}, wx[2] = {1.f - fx, fx};

  float cw[8];
  int   ci[8];
  int tc = 0;
#pragma unroll
  for (int cz = 0; cz < 2; ++cz) {
    const float zc = z0f + (float)cz;
    const bool okz = (zc >= 0.f) && (zc <= 1.f);
    const int zi = (int)fminf(fmaxf(zc, 0.f), 1.f);
#pragma unroll
    for (int cy = 0; cy < 2; ++cy) {
      const float yc = y0f + (float)cy;
      const bool oky = (yc >= 0.f) && (yc <= 7.f);
      const int yi = (int)fminf(fmaxf(yc, 0.f), 7.f);
#pragma unroll
      for (int cx = 0; cx < 2; ++cx) {
        const float xc = x0f + (float)cx;
        const bool okx = (xc >= 0.f) && (xc <= 7.f);
        const int xi = (int)fminf(fmaxf(xc, 0.f), 7.f);
        cw[tc] = (okz && oky && okx) ? wz[cz] * wy[cy] * wx[cx] : 0.f;
        ci[tc] = zi * 64 + yi * 8 + xi;
        ++tc;
      }
    }
  }

  float* outp = col + ((long)b * 32 + pos) * 4096 + (g * 64) * 16 + k;
  const float* xb = x + ((long)b * 256 + g * 64) * 128;   // D*H*W = 128
  for (int cg = 0; cg < 64; ++cg) {
    const float* xc = xb + cg * 128;
    float v = 0.f;
#pragma unroll
    for (int q = 0; q < 8; ++q) v += cw[q] * xc[ci[q]];
    outp[cg * 16] = v;
  }
}

// ---------------------------------------------------------------------------
// Host-side launchers
// ---------------------------------------------------------------------------
static inline void launch_conv(hipStream_t s, const float* x, const float* w, float* y,
                               int B, int C, int D, int H, int W,
                               int O, int kd, int kh, int kw,
                               int sd, int sh, int sw, int pd, int ph, int pw,
                               int fuse) {
  ConvP p{};
  p.x = x; p.w = w; p.col = nullptr; p.y = y;
  p.B = B; p.C = C; p.D = D; p.H = H; p.W = W;
  p.O = O; p.kd = kd; p.kh = kh; p.kw = kw;
  p.sd = sd; p.sh = sh; p.sw = sw; p.pd = pd; p.ph = ph; p.pw = pw;
  p.Do = (D + 2 * pd - kd) / sd + 1;
  p.Ho = (H + 2 * ph - kh) / sh + 1;
  p.Wo = (W + 2 * pw - kw) / sw + 1;
  p.K = C * kd * kh * kw;
  p.Nsp = B * p.Do * p.Ho * p.Wo;
  p.tilesO = O / 32;        // exact for all layers in this net
  p.tilesN = p.Nsp / 64;    // exact for all layers in this net
  p.fuseLrelu = fuse; p.precol = 0;
  const int tiles = p.tilesO * p.tilesN;
  conv_wmma_kernel<<<(tiles + 3) / 4, 128, 0, s>>>(p);
}

static inline void launch_gemm_col(hipStream_t s, const float* col, const float* w, float* y,
                                   int B, int O, int K, int Do, int Ho, int Wo) {
  ConvP p{};
  p.x = nullptr; p.w = w; p.col = col; p.y = y;
  p.B = B; p.C = 1; p.D = 1; p.H = 1; p.W = 1;
  p.O = O; p.kd = 1; p.kh = 1; p.kw = 1;
  p.sd = 1; p.sh = 1; p.sw = 1; p.pd = 0; p.ph = 0; p.pw = 0;
  p.Do = Do; p.Ho = Ho; p.Wo = Wo;
  p.K = K;
  p.Nsp = B * Do * Ho * Wo;
  p.tilesO = O / 32;
  p.tilesN = p.Nsp / 64;
  p.fuseLrelu = 0; p.precol = 1;
  const int tiles = p.tilesO * p.tilesN;
  conv_wmma_kernel<<<(tiles + 3) / 4, 128, 0, s>>>(p);
}

extern "C" void kernel_launch(void* const* d_in, const int* in_sizes, int n_in,
                              void* d_out, int out_size, void* d_ws, size_t ws_size,
                              hipStream_t stream) {
  const float* x0   = (const float*)d_in[0];
  const float* w1   = (const float*)d_in[1];
  const float* w2   = (const float*)d_in[2];
  const float* w3   = (const float*)d_in[3];
  const float* w4   = (const float*)d_in[4];
  const float* woff = (const float*)d_in[5];
  const float* w5   = (const float*)d_in[6];
  const float* w6   = (const float*)d_in[7];
  const float* g2 = (const float*)d_in[8],  *b2 = (const float*)d_in[9];
  const float* g3 = (const float*)d_in[10], *b3 = (const float*)d_in[11];
  const float* g4 = (const float*)d_in[12], *b4 = (const float*)d_in[13];
  const float* g5 = (const float*)d_in[14], *b5 = (const float*)d_in[15];
  const float* g6 = (const float*)d_in[16], *b6 = (const float*)d_in[17];

  float* out = (float*)d_out;
  float* e1 = out;                    // [64, 32,4,64,64]
  float* e2 = e1 + 33554432L;         // [64, 64,3,32,32]
  float* e3 = e2 + 12582912L;         // [64,128,2,16,16]
  float* e4 = e3 + 4194304L;          // [64,256,2, 8, 8]
  float* e5 = e4 + 2097152L;          // [64,512,2, 4, 4]
  float* e6 = e5 + 1048576L;          // [64,512,2, 1, 1]

  float* ws    = (float*)d_ws;
  float* preBN = ws;                  // max 12,582,912 floats (conv2 pre-BN)
  float* offb  = preBN + 12582912L;   //   393,216 floats
  float* colb  = offb  + 393216L;     // 8,388,608 floats (deform im2col)
  float* meanb = colb  + 8388608L;    //       512 floats
  float* varb  = meanb + 512L;        //       512 floats

  // ---- Layer 1: conv + LeakyReLU (no BN in forward) ----
  launch_conv(stream, x0, w1, e1, 64, 3, 8, 128, 128, 32, 4, 4, 4, 2, 2, 2, 1, 1, 1, 1);

  // ---- Layer 2 ----
  launch_conv(stream, e1, w2, preBN, 64, 32, 4, 64, 64, 64, 2, 4, 4, 1, 2, 2, 0, 1, 1, 0);
  bn_reduce_kernel<<<64, 256, 0, stream>>>(preBN, meanb, varb, 64, 64, 3 * 32 * 32);
  {
    long tot = 64L * 64 * 3 * 32 * 32;
    bn_apply_lrelu_kernel<<<(int)((tot + 255) / 256), 256, 0, stream>>>(
        preBN, e2, meanb, varb, g2, b2, 64, 3 * 32 * 32, tot);
  }

  // ---- Layer 3 ----
  launch_conv(stream, e2, w3, preBN, 64, 64, 3, 32, 32, 128, 2, 4, 4, 1, 2, 2, 0, 1, 1, 0);
  bn_reduce_kernel<<<128, 256, 0, stream>>>(preBN, meanb, varb, 128, 64, 2 * 16 * 16);
  {
    long tot = 64L * 128 * 2 * 16 * 16;
    bn_apply_lrelu_kernel<<<(int)((tot + 255) / 256), 256, 0, stream>>>(
        preBN, e3, meanb, varb, g3, b3, 128, 2 * 16 * 16, tot);
  }

  // ---- Layer 4 ----
  launch_conv(stream, e3, w4, preBN, 64, 128, 2, 16, 16, 256, 1, 4, 4, 1, 2, 2, 0, 1, 1, 0);
  bn_reduce_kernel<<<256, 256, 0, stream>>>(preBN, meanb, varb, 256, 64, 2 * 8 * 8);
  {
    long tot = 64L * 256 * 2 * 8 * 8;
    bn_apply_lrelu_kernel<<<(int)((tot + 255) / 256), 256, 0, stream>>>(
        preBN, e4, meanb, varb, g4, b4, 256, 2 * 8 * 8, tot);
  }

  // ---- Offset conv (plain conv, no BN/activation) ----
  launch_conv(stream, e4, woff, offb, 64, 256, 2, 8, 8, 192, 1, 4, 4, 1, 2, 2, 0, 1, 1, 0);

  // ---- Deformable conv: trilinear sample -> im2col, then WMMA GEMM ----
  {
    const int nthreads = 64 * 4 * 16 * 32;
    deform_sample_kernel<<<(nthreads + 255) / 256, 256, 0, stream>>>(e4, offb, colb);
  }
  launch_gemm_col(stream, colb, w5, preBN, 64, 512, 4096, 2, 4, 4);
  bn_reduce_kernel<<<512, 256, 0, stream>>>(preBN, meanb, varb, 512, 64, 2 * 4 * 4);
  {
    long tot = 64L * 512 * 2 * 4 * 4;
    bn_apply_lrelu_kernel<<<(int)((tot + 255) / 256), 256, 0, stream>>>(
        preBN, e5, meanb, varb, g5, b5, 512, 2 * 4 * 4, tot);
  }

  // ---- Layer 6 ----
  launch_conv(stream, e5, w6, preBN, 64, 512, 2, 4, 4, 512, 1, 4, 4, 1, 1, 1, 0, 0, 0, 0);
  bn_reduce_kernel<<<512, 256, 0, stream>>>(preBN, meanb, varb, 512, 64, 2);
  {
    long tot = 64L * 512 * 2;
    bn_apply_lrelu_kernel<<<(int)((tot + 255) / 256), 256, 0, stream>>>(
        preBN, e6, meanb, varb, g6, b6, 512, 2, tot);
  }
}
